// GNNModel_56023553409388
// MI455X (gfx1250) — matrix operand order repaired and checked by reading, not compiled, not run on hardware
//
#include <hip/hip_runtime.h>

#define SLOPE 0.01f
#define EPSBN 1e-5f

typedef __attribute__((ext_vector_type(16))) __bf16 v16bfx;
typedef __attribute__((ext_vector_type(8)))  __bf16 v8bfx;
typedef __attribute__((ext_vector_type(8)))  float  v8fx;

__device__ __forceinline__ float lrelu(float v) { return v >= 0.f ? v : SLOPE * v; }

// ---------------------------------------------------------------------------
// Prep kernels
// ---------------------------------------------------------------------------
__global__ void conv_pad_bf16(const float* __restrict__ src, __bf16* __restrict__ dst,
                              int N, int Kin, int Kpad) {
    int i = blockIdx.x * blockDim.x + threadIdx.x;
    if (i >= N * Kpad) return;
    int n = i / Kpad, k = i - n * Kpad;
    dst[i] = (k < Kin) ? (__bf16)src[n * Kin + k] : (__bf16)0.0f;
}

// s = g * rsqrt(v+eps);  t = (b_lin - m)*s + be   (b_lin optional)
__global__ void bn_fold(const float* __restrict__ g, const float* __restrict__ be,
                        const float* __restrict__ m, const float* __restrict__ v,
                        const float* __restrict__ blin,
                        float* __restrict__ s, float* __restrict__ t, int n) {
    int i = blockIdx.x * blockDim.x + threadIdx.x;
    if (i >= n) return;
    float si = g[i] * rsqrtf(v[i] + EPSBN);
    float bl = blin ? blin[i] : 0.0f;
    s[i] = si;
    t[i] = (bl - m[i]) * si + be[i];
}

// out = mean_n( h . W4[n] + b4[n] )  ->  h . w4sum + b4mean
__global__ void w4_fold(const float* __restrict__ W4, const float* __restrict__ b4,
                        float* __restrict__ w4sum, float* __restrict__ b4m) {
    int k = threadIdx.x; // 128 threads
    float a = 0.f;
#pragma unroll
    for (int n = 0; n < 8; n++) a += W4[n * 128 + k];
    w4sum[k] = a * 0.125f;
    if (k == 0) {
        float s = 0.f;
#pragma unroll
        for (int n = 0; n < 8; n++) s += b4[n];
        *b4m = s * 0.125f;
    }
}

// ---------------------------------------------------------------------------
// GNN front: one thread per graph. Weights staged in LDS (broadcast reads).
// Edge indices are batch-invariant -> readfirstlane -> uniform branches, so
// node gather/scatter uses only compile-time register indexing.
// ---------------------------------------------------------------------------
// LDS float layout offsets
#define SM_WNODE 0        // 96
#define SM_BNODE 96       // 16
#define SM_WEDGE 112      // 8
#define SM_BEDGE 120      // 4
#define SM_MSGW  124      // 4*16*36 = 2304
#define SM_MSGB  2428     // 64
#define SM_UPDW  2492     // 4*16*32 = 2048
#define SM_UPDB  4540     // 64
#define SM_BNS   4604     // 20
#define SM_BNT   4624     // 20
#define SM_POOLW 4644     // 1280
#define SM_POOLB 5924     // 16
#define SM_TOT   5940

__global__ __launch_bounds__(256) void gnn_front(
    const float* __restrict__ node_feat, const float* __restrict__ edge_feat,
    const int* __restrict__ edge_index,
    const float* __restrict__ W_node, const float* __restrict__ b_node,
    const float* __restrict__ W_edge, const float* __restrict__ b_edge,
    const float* __restrict__ msg_W, const float* __restrict__ msg_b,
    const float* __restrict__ upd_W, const float* __restrict__ upd_b,
    const float* __restrict__ bns, const float* __restrict__ bnt,
    const float* __restrict__ pool_W, const float* __restrict__ pool_b,
    __bf16* __restrict__ p_out, int B)
{
    __shared__ float sm[SM_TOT];
    __shared__ int sEI[40];
    int tid = threadIdx.x;
    for (int i = tid; i < 96;   i += 256) sm[SM_WNODE + i] = W_node[i];
    for (int i = tid; i < 16;   i += 256) sm[SM_BNODE + i] = b_node[i];
    for (int i = tid; i < 8;    i += 256) sm[SM_WEDGE + i] = W_edge[i];
    for (int i = tid; i < 4;    i += 256) sm[SM_BEDGE + i] = b_edge[i];
    for (int i = tid; i < 2304; i += 256) sm[SM_MSGW + i] = msg_W[i];
    for (int i = tid; i < 64;   i += 256) sm[SM_MSGB + i] = msg_b[i];
    for (int i = tid; i < 2048; i += 256) sm[SM_UPDW + i] = upd_W[i];
    for (int i = tid; i < 64;   i += 256) sm[SM_UPDB + i] = upd_b[i];
    for (int i = tid; i < 20;   i += 256) sm[SM_BNS + i]  = bns[i];
    for (int i = tid; i < 20;   i += 256) sm[SM_BNT + i]  = bnt[i];
    for (int i = tid; i < 1280; i += 256) sm[SM_POOLW + i] = pool_W[i];
    for (int i = tid; i < 16;   i += 256) sm[SM_POOLB + i] = pool_b[i];
    for (int i = tid; i < 40;   i += 256) sEI[i] = edge_index[i];
    __syncthreads();

    int b = blockIdx.x * 256 + tid;
    if (b >= B) return;

    // node projection: x[n][k]
    float x[5][16];
    {
        const float* nf = node_feat + (size_t)b * 30;
#pragma unroll
        for (int n = 0; n < 5; n++) {
            float nreg[6];
#pragma unroll
            for (int f = 0; f < 6; f++) nreg[f] = nf[n * 6 + f];
#pragma unroll
            for (int k = 0; k < 16; k++) {
                float a = sm[SM_BNODE + k];
#pragma unroll
                for (int f = 0; f < 6; f++) a += sm[SM_WNODE + k * 6 + f] * nreg[f];
                x[n][k] = a;
            }
        }
    }

    const float* ef = edge_feat + (size_t)b * 40;

#pragma unroll 1
    for (int l = 0; l < 4; l++) {
        const float* mw = &sm[SM_MSGW + l * 576];
        const float* mb = &sm[SM_MSGB + l * 16];
        float aggr[5][16];
#pragma unroll
        for (int n = 0; n < 5; n++)
#pragma unroll
            for (int k = 0; k < 16; k++) aggr[n][k] = 0.f;

#pragma unroll 1
        for (int e = 0; e < 20; e++) {
            int se = __builtin_amdgcn_readfirstlane(sEI[e]);       // src (uniform)
            int de = __builtin_amdgcn_readfirstlane(sEI[20 + e]);  // dst (uniform)
            float e0 = ef[2 * e], e1 = ef[2 * e + 1];
            float ea[4];
#pragma unroll
            for (int k = 0; k < 4; k++)
                ea[k] = sm[SM_BEDGE + k] + sm[SM_WEDGE + 2 * k] * e0 + sm[SM_WEDGE + 2 * k + 1] * e1;
            float hi[16], hj[16];
#pragma unroll
            for (int n = 0; n < 5; n++) {
                if (de == n) {
#pragma unroll
                    for (int k = 0; k < 16; k++) hi[k] = x[n][k];
                }
                if (se == n) {
#pragma unroll
                    for (int k = 0; k < 16; k++) hj[k] = x[n][k];
                }
            }
            float msg[16];
#pragma unroll
            for (int k = 0; k < 16; k++) {
                float a = mb[k];
                const float* w = mw + k * 36;
#pragma unroll
                for (int c = 0; c < 16; c++) a += w[c] * hi[c];
#pragma unroll
                for (int c = 0; c < 16; c++) a += w[16 + c] * hj[c];
#pragma unroll
                for (int c = 0; c < 4; c++) a += w[32 + c] * ea[c];
                msg[k] = lrelu(a);
            }
#pragma unroll
            for (int n = 0; n < 5; n++) {
                if (de == n) {
#pragma unroll
                    for (int k = 0; k < 16; k++) aggr[n][k] += msg[k];
                }
            }
        }

        const float* uw = &sm[SM_UPDW + l * 512];
        const float* ub = &sm[SM_UPDB + l * 16];
#pragma unroll
        for (int n = 0; n < 5; n++) {
            float s_ = sm[SM_BNS + l * 5 + n];
            float t_ = sm[SM_BNT + l * 5 + n];
            float tmp[16];
#pragma unroll
            for (int k = 0; k < 16; k++) {
                float a = ub[k];
                const float* w = uw + k * 32;
#pragma unroll
                for (int c = 0; c < 16; c++) a += w[c] * x[n][c];
#pragma unroll
                for (int c = 0; c < 16; c++) a += w[16 + c] * aggr[n][c];
                a = a * s_ + t_;
                tmp[k] = lrelu(a);
            }
#pragma unroll
            for (int k = 0; k < 16; k++) x[n][k] += tmp[k];
        }
    }

    // pool -> p (16), stored bf16 with K padded to 32 for WMMA
    __bf16* po = p_out + (size_t)b * 32;
#pragma unroll
    for (int k = 0; k < 16; k++) {
        float a = sm[SM_POOLB + k];
        const float* w = &sm[SM_POOLW + k * 80];
#pragma unroll
        for (int n = 0; n < 5; n++)
#pragma unroll
            for (int f = 0; f < 16; f++) a += w[n * 16 + f] * x[n][f];
        po[k] = (__bf16)lrelu(a);
    }
#pragma unroll
    for (int k = 16; k < 32; k++) po[k] = (__bf16)0.0f;
}

// ---------------------------------------------------------------------------
// Fused WMMA MLP: per WG 64 rows, 8 waves, all activations in LDS (208 KB).
// A frag (16-bit 16x32): lane L, row = L&15, VGPR0-3 = K[koff,koff+8),
// VGPR4-7 = K[16+koff,..), koff = (L>=16)?8:0.
// B frag from row-major bf16 weights W(N,K): lane L, n = L&15,
// same K layout as A: VGPR0-3 = K[koff,koff+8), VGPR4-7 = K[16+koff,..).
// D frag (f32 16x16): lane L, n=L&15, element v -> m = v + (L>=16?8:0).
// ---------------------------------------------------------------------------
template <int K, int N, int MTILES>
__device__ __forceinline__ void gemm_layer(const __bf16* __restrict__ A,
                                           const __bf16* __restrict__ Wb,
                                           const float* __restrict__ s,
                                           const float* __restrict__ t,
                                           __bf16* __restrict__ D,
                                           int wave, int lane)
{
    const int NT = N / 16;
    const int tiles = MTILES * NT;
    const int nlane = lane & 15;
    const int hi = lane >> 4;
    const int koffA = hi * 8;
    const int koffB = hi * 8;
    const int KS = K / 32;
    for (int t0 = wave; t0 < tiles; t0 += 8) {
        int mt = t0 / NT, nt = t0 - mt * NT;
        const __bf16* arow = A + (size_t)(mt * 16 + nlane) * K;
        const __bf16* brow = Wb + (size_t)(nt * 16 + nlane) * K + koffB;
        v8fx c = {};
#pragma unroll 4
        for (int ks = 0; ks < KS; ks++) {
            v16bfx a, b;
            const __bf16* ap = arow + ks * 32;
            *((v8bfx*)&a)     = *(const v8bfx*)(ap + koffA);
            *((v8bfx*)&a + 1) = *(const v8bfx*)(ap + 16 + koffA);
            const __bf16* bp = brow + ks * 32;
            *((v8bfx*)&b)     = *(const v8bfx*)(bp);
            *((v8bfx*)&b + 1) = *(const v8bfx*)(bp + 16);
            c = __builtin_amdgcn_wmma_f32_16x16x32_bf16(false, a, false, b,
                                                        (short)0, c, false, false);
        }
        float ss = s[nt * 16 + nlane];
        float tt = t[nt * 16 + nlane];
#pragma unroll
        for (int v = 0; v < 8; v++) {
            float val = lrelu(c[v] * ss + tt);
            D[(size_t)(mt * 16 + hi * 8 + v) * N + nt * 16 + nlane] = (__bf16)val;
        }
    }
}

#define MLP_LDS_BYTES (64 * 1024 * 2 + 64 * 512 * 2 + 64 * 128 * 2) // 212992 < 320KB/WG

__global__ __launch_bounds__(256) void mlp_kernel(
    const __bf16* __restrict__ P, const __bf16* __restrict__ W1b,
    const __bf16* __restrict__ W2b, const __bf16* __restrict__ W3b,
    const float* __restrict__ s1, const float* __restrict__ t1,
    const float* __restrict__ s2, const float* __restrict__ t2,
    const float* __restrict__ s3, const float* __restrict__ t3,
    const float* __restrict__ w4sum, const float* __restrict__ b4m,
    float* __restrict__ out)
{
    extern __shared__ __align__(16) char smem[];
    __bf16* h1 = (__bf16*)smem;                               // 64 x 1024
    __bf16* h2 = (__bf16*)(smem + 64 * 1024 * 2);             // 64 x 512
    __bf16* h3 = (__bf16*)(smem + 64 * 1024 * 2 + 64 * 512 * 2); // 64 x 128

    int tid = threadIdx.x;
    int wave = tid >> 5, lane = tid & 31;
    int row0 = blockIdx.x * 64;

    // L1: (64x32) @ (1024x32)^T -> h1, BN+lrelu fused
    gemm_layer<32, 1024, 4>(P + (size_t)row0 * 32, W1b, s1, t1, h1, wave, lane);
    __syncthreads();
    // L2: (64x1024) @ (512x1024)^T -> h2   (dominant GEMM)
    gemm_layer<1024, 512, 4>(h1, W2b, s2, t2, h2, wave, lane);
    __syncthreads();
    // L3: (64x512) @ (128x512)^T -> h3
    gemm_layer<512, 128, 4>(h2, W3b, s3, t3, h3, wave, lane);
    __syncthreads();

    // L4 + mean folded: out[m] = h3[m,:] . w4sum + b4mean
    if (tid < 64) {
        float acc = *b4m;
        const __bf16* hr = h3 + tid * 128;
#pragma unroll 8
        for (int k = 0; k < 128; k++) acc += (float)hr[k] * w4sum[k];
        out[row0 + tid] = acc;
    }
}

// ---------------------------------------------------------------------------
extern "C" void kernel_launch(void* const* d_in, const int* in_sizes, int n_in,
                              void* d_out, int out_size, void* d_ws, size_t ws_size,
                              hipStream_t stream) {
    (void)n_in; (void)out_size; (void)ws_size;
    const float* node_feat = (const float*)d_in[0];
    const float* edge_feat = (const float*)d_in[1];
    const int*   edge_index = (const int*)d_in[2];
    const float* W_node = (const float*)d_in[3];
    const float* b_node = (const float*)d_in[4];
    const float* W_edge = (const float*)d_in[5];
    const float* b_edge = (const float*)d_in[6];
    const float* msg_W = (const float*)d_in[7];
    const float* msg_b = (const float*)d_in[8];
    const float* upd_W = (const float*)d_in[9];
    const float* upd_b = (const float*)d_in[10];
    const float* bn_g = (const float*)d_in[11];
    const float* bn_b = (const float*)d_in[12];
    const float* bn_m = (const float*)d_in[13];
    const float* bn_v = (const float*)d_in[14];
    const float* pool_W = (const float*)d_in[15];
    const float* pool_b = (const float*)d_in[16];
    const float* W1 = (const float*)d_in[17];
    const float* b1 = (const float*)d_in[18];
    const float* g1 = (const float*)d_in[19];
    const float* be1 = (const float*)d_in[20];
    const float* m1 = (const float*)d_in[21];
    const float* v1 = (const float*)d_in[22];
    const float* W2 = (const float*)d_in[23];
    const float* b2 = (const float*)d_in[24];
    const float* g2 = (const float*)d_in[25];
    const float* be2 = (const float*)d_in[26];
    const float* m2 = (const float*)d_in[27];
    const float* v2 = (const float*)d_in[28];
    const float* W3 = (const float*)d_in[29];
    const float* b3 = (const float*)d_in[30];
    const float* g3 = (const float*)d_in[31];
    const float* be3 = (const float*)d_in[32];
    const float* m3 = (const float*)d_in[33];
    const float* v3 = (const float*)d_in[34];
    const float* W4 = (const float*)d_in[35];
    const float* b4 = (const float*)d_in[36];

    const int B = in_sizes[0] / 30; // node_feat = B*5*6

    char* ws = (char*)d_ws;
    size_t off = 0;
    auto alloc = [&](size_t bytes) {
        void* p = ws + off;
        off += (bytes + 255) & ~(size_t)255;
        return p;
    };
    __bf16* p_bf  = (__bf16*)alloc((size_t)B * 32 * 2);
    __bf16* W1b   = (__bf16*)alloc(1024 * 32 * 2);
    __bf16* W2b   = (__bf16*)alloc(512 * 1024 * 2);
    __bf16* W3b   = (__bf16*)alloc(128 * 512 * 2);
    float*  s1    = (float*)alloc(1024 * 4);
    float*  t1    = (float*)alloc(1024 * 4);
    float*  s2    = (float*)alloc(512 * 4);
    float*  t2    = (float*)alloc(512 * 4);
    float*  s3    = (float*)alloc(128 * 4);
    float*  t3    = (float*)alloc(128 * 4);
    float*  w4sum = (float*)alloc(128 * 4);
    float*  b4m   = (float*)alloc(4);
    float*  bns   = (float*)alloc(20 * 4);
    float*  bnt   = (float*)alloc(20 * 4);

    // --- prep ---
    conv_pad_bf16<<<(1024 * 32 + 255) / 256, 256, 0, stream>>>(W1, W1b, 1024, 16, 32);
    conv_pad_bf16<<<(512 * 1024 + 255) / 256, 256, 0, stream>>>(W2, W2b, 512, 1024, 1024);
    conv_pad_bf16<<<(128 * 512 + 255) / 256, 256, 0, stream>>>(W3, W3b, 128, 512, 512);
    bn_fold<<<4, 256, 0, stream>>>(g1, be1, m1, v1, b1, s1, t1, 1024);
    bn_fold<<<2, 256, 0, stream>>>(g2, be2, m2, v2, b2, s2, t2, 512);
    bn_fold<<<1, 128, 0, stream>>>(g3, be3, m3, v3, b3, s3, t3, 128);
    bn_fold<<<1, 32, 0, stream>>>(bn_g, bn_b, bn_m, bn_v, nullptr, bns, bnt, 20);
    w4_fold<<<1, 128, 0, stream>>>(W4, b4, w4sum, b4m);

    // --- GNN front ---
    gnn_front<<<(B + 255) / 256, 256, 0, stream>>>(
        node_feat, edge_feat, edge_index, W_node, b_node, W_edge, b_edge,
        msg_W, msg_b, upd_W, upd_b, bns, bnt, pool_W, pool_b, p_bf, B);

    // --- fused WMMA MLP ---
    hipFuncSetAttribute((const void*)mlp_kernel,
                        hipFuncAttributeMaxDynamicSharedMemorySize, MLP_LDS_BYTES);
    mlp_kernel<<<B / 64, 256, MLP_LDS_BYTES, stream>>>(
        p_bf, W1b, W2b, W3b, s1, t1, s2, t2, s3, t3, w4sum, b4m, (float*)d_out);
}
